// SentenceRetrival_77077483094304
// MI455X (gfx1250) — matrix operand order repaired and checked by reading, not compiled, not run on hardware
//
#include <hip/hip_runtime.h>
#include <hip/hip_bf16.h>

#define NUSR 100000
#define NITM 50000
#define DIM  128
#define CLS  5
#define NE   200000
#define LL   10

typedef __attribute__((ext_vector_type(16))) __bf16 v16bf;
typedef __attribute__((ext_vector_type(8)))  __bf16 v8bf;
typedef __attribute__((ext_vector_type(8)))  float  v8f;
typedef __attribute__((ext_vector_type(4)))  float  v4f;

#define WAVES 8
#define TPB   (WAVES * 32)
#define LD1   264           // bf16 K-stride for [128][256] transposed weights (+8 pad)
#define LD2   136           // bf16 K-stride for [128][128] transposed weights (+8 pad)
#define SCR_LD 132          // f32 row stride of 16x128 per-wave scratch tile (+4 pad)

#define OFF_W1R 0
#define OFF_W2R (OFF_W1R + 128 * LD1 * 2)
#define OFF_W1T (OFF_W2R + 128 * LD2 * 2)
#define OFF_W2T (OFF_W1T + 128 * LD1 * 2)
#define OFF_WP  (OFF_W2T + 128 * LD2 * 2)
#define OFF_SCR (OFF_WP + DIM * CLS * 4)
#define SCR_BYTES (16 * SCR_LD * 4)
#define SMEM_BYTES (OFF_SCR + WAVES * SCR_BYTES)   // 274944 B < 320 KB

__device__ __forceinline__ v8f wmma_bf16(v16bf a, v16bf b, v8f c) {
    return __builtin_amdgcn_wmma_f32_16x16x32_bf16(false, a, false, b,
                                                   (short)0, c, false, false);
}

// B fragment (32x16 K x N, bf16): lane n = lane&15 selects column,
// K = kb*32 + 8*(lane>=16) + i + 8*(i>=8), i=0..15, in transposed [N][K] LDS.
__device__ __forceinline__ v16bf load_b_frag(const __bf16* wT, int ldk,
                                             int nt, int kb, int lane) {
    int n  = nt * 16 + (lane & 15);
    int k0 = kb * 32 + ((lane & 16) ? 8 : 0);
    const __bf16* p = wT + n * ldk + k0;
    v8bf lo = *(const v8bf*)(p);
    v8bf hi = *(const v8bf*)(p + 16);
    return __builtin_shufflevector(lo, hi, 0,1,2,3,4,5,6,7,8,9,10,11,12,13,14,15);
}

// A fragment (16x32 M x K, bf16) from a concat of two global f32 rows.
// lane row M = lane&15; element i: k = k0 + i + 8*(i>=8), k0 = kb*32 + 8*(lane>=16)
// rowA covers k in [0,128), rowB covers [128,256)
__device__ __forceinline__ v16bf load_a_concat(const float* rowA, const float* rowB, int k0) {
    v16bf r;
    #pragma unroll
    for (int c = 0; c < 2; ++c) {
        int kk = k0 + c * 16;
        const float* p = (kk < DIM) ? (rowA + kk) : (rowB + (kk - DIM));
        v4f x0 = *(const v4f*)(p);
        v4f x1 = *(const v4f*)(p + 4);
        #pragma unroll
        for (int i = 0; i < 4; ++i) {
            r[c * 8 + i]     = (__bf16)x0[i];
            r[c * 8 + 4 + i] = (__bf16)x1[i];
        }
    }
    return r;
}

// A fragment from a contiguous f32 LDS row (layer-2 hidden tile).
__device__ __forceinline__ v16bf load_a_lds(const float* row, int k0) {
    v16bf r;
    #pragma unroll
    for (int c = 0; c < 2; ++c) {
        const float* p = row + k0 + c * 16;
        #pragma unroll
        for (int i = 0; i < 8; ++i) r[c * 8 + i] = (__bf16)p[i];
    }
    return r;
}

__global__ void zero_loss_kernel(float* out) {
    if (threadIdx.x == 0) {
        out[(size_t)NE * CLS]     = 0.0f;
        out[(size_t)NE * CLS + 1] = 0.0f;
    }
}

__global__ void finish_loss_kernel(float* out) {
    if (threadIdx.x == 0) out[(size_t)NE * CLS + 1] = out[(size_t)NE * CLS];
}

__global__ void __launch_bounds__(TPB, 1)
fused_retrieval_kernel(const float* __restrict__ urf, const float* __restrict__ irf,
                       const float* __restrict__ utf, const float* __restrict__ itf,
                       const float* __restrict__ w1r, const float* __restrict__ w2r,
                       const float* __restrict__ w1t, const float* __restrict__ w2t,
                       const float* __restrict__ wp,  const float* __restrict__ emb,
                       const int* __restrict__ src,   const int* __restrict__ dst,
                       const int* __restrict__ sid,   const int* __restrict__ nid,
                       float* __restrict__ out) {
    extern __shared__ char smem[];
    __bf16* w1rT = (__bf16*)(smem + OFF_W1R);
    __bf16* w2rT = (__bf16*)(smem + OFF_W2R);
    __bf16* w1tT = (__bf16*)(smem + OFF_W1T);
    __bf16* w2tT = (__bf16*)(smem + OFF_W2T);
    float*  wpl  = (float*)(smem + OFF_WP);

    const int tid = threadIdx.x;

    // ---- stage weights to LDS (bf16, transposed [N][K]) ----
    for (int idx = tid; idx < 2 * DIM * DIM; idx += TPB) {   // [256][128]
        int k = idx >> 7, n = idx & 127;
        w1rT[n * LD1 + k] = (__bf16)w1r[idx];
        w1tT[n * LD1 + k] = (__bf16)w1t[idx];
    }
    for (int idx = tid; idx < DIM * DIM; idx += TPB) {       // [128][128]
        int k = idx >> 7, n = idx & 127;
        w2rT[n * LD2 + k] = (__bf16)w2r[idx];
        w2tT[n * LD2 + k] = (__bf16)w2t[idx];
    }
    for (int idx = tid; idx < DIM * CLS; idx += TPB) wpl[idx] = wp[idx];
    __syncthreads();

    const int wave = tid >> 5;
    const int lane = tid & 31;
    const int mrow = lane & 15;   // edge row within tile (A/C row, also B/C col)
    const int half = lane >> 4;
    float* scr = (float*)(smem + OFF_SCR + wave * SCR_BYTES);

    const int numTiles = (NE + 15) / 16;
    for (int t = blockIdx.x * WAVES + wave; t < numTiles; t += gridDim.x * WAVES) {
        int e  = t * 16 + mrow;
        int ec = (e < NE) ? e : (NE - 1);
        int su = src[ec], di = dst[ec];
        const float* ur = urf + (size_t)su * DIM;
        const float* ir = irf + (size_t)di * DIM;
        const float* ut = utf + (size_t)su * DIM;
        const float* it = itf + (size_t)di * DIM;

        // ---------- rating MLP layer 1: relu(rf @ w1r), K=256 ----------
        v8f acc[8];
        #pragma unroll
        for (int nt = 0; nt < 8; ++nt)
            #pragma unroll
            for (int r = 0; r < 8; ++r) acc[nt][r] = 0.0f;

        for (int kb = 0; kb < 8; ++kb) {
            v16bf a = load_a_concat(ur, ir, kb * 32 + half * 8);
            #pragma unroll
            for (int nt = 0; nt < 8; ++nt)
                acc[nt] = wmma_bf16(a, load_b_frag(w1rT, LD1, nt, kb, lane), acc[nt]);
        }
        // C layout: col = lane&15, row = r + 8*half -> store transposed tile to LDS
        #pragma unroll
        for (int nt = 0; nt < 8; ++nt)
            #pragma unroll
            for (int r = 0; r < 8; ++r)
                scr[(r + 8 * half) * SCR_LD + nt * 16 + mrow] = fmaxf(acc[nt][r], 0.0f);

        // ---------- rating layer 2: rh = h @ w2r, K=128 ----------
        v8f rh[8];
        #pragma unroll
        for (int nt = 0; nt < 8; ++nt)
            #pragma unroll
            for (int r = 0; r < 8; ++r) rh[nt][r] = 0.0f;

        for (int kb = 0; kb < 4; ++kb) {
            v16bf a = load_a_lds(scr + mrow * SCR_LD, kb * 32 + half * 8);
            #pragma unroll
            for (int nt = 0; nt < 8; ++nt)
                rh[nt] = wmma_bf16(a, load_b_frag(w2rT, LD2, nt, kb, lane), rh[nt]);
        }
        #pragma unroll
        for (int nt = 0; nt < 8; ++nt)
            #pragma unroll
            for (int r = 0; r < 8; ++r)
                scr[(r + 8 * half) * SCR_LD + nt * 16 + mrow] = rh[nt][r];

        // ---------- pr = rh @ wp  (16 x 5 per tile, VALU) ----------
        for (int p = lane; p < 16 * CLS; p += 32) {
            int mm = p / CLS, c = p % CLS;
            float s = 0.0f;
            for (int n = 0; n < DIM; ++n) s += scr[mm * SCR_LD + n] * wpl[n * CLS + c];
            int ee = t * 16 + mm;
            if (ee < NE) out[(size_t)ee * CLS + c] = s;
        }

        // ---------- topic MLP layer 1: relu(tf @ w1t) ----------
        #pragma unroll
        for (int nt = 0; nt < 8; ++nt)
            #pragma unroll
            for (int r = 0; r < 8; ++r) acc[nt][r] = 0.0f;

        for (int kb = 0; kb < 8; ++kb) {
            v16bf a = load_a_concat(ut, it, kb * 32 + half * 8);
            #pragma unroll
            for (int nt = 0; nt < 8; ++nt)
                acc[nt] = wmma_bf16(a, load_b_frag(w1tT, LD1, nt, kb, lane), acc[nt]);
        }
        #pragma unroll
        for (int nt = 0; nt < 8; ++nt)
            #pragma unroll
            for (int r = 0; r < 8; ++r)
                scr[(r + 8 * half) * SCR_LD + nt * 16 + mrow] = fmaxf(acc[nt][r], 0.0f);

        // ---------- topic layer 2 with residual: th = h_t @ w2t + rh (C = rh) ----------
        for (int kb = 0; kb < 4; ++kb) {
            v16bf a = load_a_lds(scr + mrow * SCR_LD, kb * 32 + half * 8);
            #pragma unroll
            for (int nt = 0; nt < 8; ++nt)
                rh[nt] = wmma_bf16(a, load_b_frag(w2tT, LD2, nt, kb, lane), rh[nt]);
        }
        #pragma unroll
        for (int nt = 0; nt < 8; ++nt)
            #pragma unroll
            for (int r = 0; r < 8; ++r)
                scr[(r + 8 * half) * SCR_LD + nt * 16 + mrow] = rh[nt][r];

        // ---------- review scoring: lane-pair per edge, 64 dims each ----------
        v4f th4[16];
        const float* throw_ = scr + mrow * SCR_LD + half * 64;
        #pragma unroll
        for (int i = 0; i < 16; ++i) th4[i] = *(const v4f*)(throw_ + i * 4);

        const int* pid = sid + (size_t)ec * LL;
        const int* ngd = nid + (size_t)ec * LL;
        float ap = 0.0f, an = 0.0f;
        int   cp = 0,    cn = 0;
        for (int j = 0; j < LL; ++j) {
            int ip = pid[j]; cp += (ip > 0);
            const float* ep = emb + (size_t)ip * DIM + half * 64;
            #pragma unroll
            for (int i = 0; i < 16; ++i) {
                v4f ev = *(const v4f*)(ep + i * 4);
                ap += th4[i][0] * ev[0] + th4[i][1] * ev[1]
                    + th4[i][2] * ev[2] + th4[i][3] * ev[3];
            }
            int in_ = ngd[j]; cn += (in_ > 0);
            const float* en = emb + (size_t)in_ * DIM + half * 64;
            #pragma unroll
            for (int i = 0; i < 16; ++i) {
                v4f ev = *(const v4f*)(en + i * 4);
                an += th4[i][0] * ev[0] + th4[i][1] * ev[1]
                    + th4[i][2] * ev[2] + th4[i][3] * ev[3];
            }
        }
        ap += __shfl_xor(ap, 16, 32);
        an += __shfl_xor(an, 16, 32);
        if (half == 0 && e < NE) {
            float ps = ap / ((float)cp + 1e-9f);
            float ns = an / ((float)cn + 1e-9f);
            float x  = ps - ns;
            float loss = (x > 0.0f) ? log1pf(__expf(-x)) : (-x + log1pf(__expf(x)));
            atomicAdd(out + (size_t)NE * CLS, loss * (1.0f / (float)NE));
        }
    }
}

extern "C" void kernel_launch(void* const* d_in, const int* in_sizes, int n_in,
                              void* d_out, int out_size, void* d_ws, size_t ws_size,
                              hipStream_t stream) {
    (void)in_sizes; (void)n_in; (void)out_size; (void)d_ws; (void)ws_size;
    const float* urf = (const float*)d_in[0];
    const float* irf = (const float*)d_in[1];
    const float* utf = (const float*)d_in[2];
    const float* itf = (const float*)d_in[3];
    const float* w1r = (const float*)d_in[4];
    const float* w2r = (const float*)d_in[5];
    const float* w1t = (const float*)d_in[6];
    const float* w2t = (const float*)d_in[7];
    const float* wp  = (const float*)d_in[8];
    const float* emb = (const float*)d_in[9];
    const int*   src = (const int*)d_in[10];
    const int*   dst = (const int*)d_in[11];
    const int*   sid = (const int*)d_in[12];
    const int*   nid = (const int*)d_in[13];
    float* out = (float*)d_out;

    (void)hipFuncSetAttribute((const void*)fused_retrieval_kernel,
                              hipFuncAttributeMaxDynamicSharedMemorySize, SMEM_BYTES);

    zero_loss_kernel<<<1, 32, 0, stream>>>(out);

    const int numTiles = (NE + 15) / 16;
    int blocks = (numTiles + WAVES - 1) / WAVES;
    if (blocks > 320) blocks = 320;   // grid-stride; ~1 big-LDS block per WGP
    if (blocks < 1) blocks = 1;
    fused_retrieval_kernel<<<blocks, TPB, SMEM_BYTES, stream>>>(
        urf, irf, utf, itf, w1r, w2r, w1t, w2t, wp, emb, src, dst, sid, nid, out);

    finish_loss_kernel<<<1, 32, 0, stream>>>(out);
}
